// CustomGaussianLayer_26182120636507
// MI455X (gfx1250) — matrix operand order repaired and checked by reading, not compiled, not run on hardware
//
#include <hip/hip_runtime.h>
#include <hip/hip_bf16.h>
#include <cstdint>

typedef __attribute__((ext_vector_type(16))) _Float16 v16h;
typedef __attribute__((ext_vector_type(8)))  _Float16 v8h;
typedef __attribute__((ext_vector_type(8)))  float    v8f;
typedef __attribute__((ext_vector_type(4)))  float    v4f;
typedef int v4i_vs __attribute__((vector_size(16)));   // matches builtin pointee

#define GAUSS_G   8
#define IN_SZ     512
#define OUT_SZ    512
#define BATCH_SZ  8192
#define K_SZ      (IN_SZ * GAUSS_G)   // 4096

#define BM 128
#define BN 128
#define BK 64
#define NT (K_SZ / BK)                // 64 K-tiles
#define LDS_STRIDE 72                 // BK + 8 pad (f16); row = 144B, 16B aligned

// ---- CDNA5 async-to-LDS path (guarded; falls back to sync copy) ----------
#if __has_builtin(__builtin_amdgcn_global_load_async_to_lds_b128) && \
    __has_builtin(__builtin_amdgcn_s_wait_asynccnt)
#define USE_ASYNC_LDS 1
#define WAIT_ASYNC(n) __builtin_amdgcn_s_wait_asynccnt(n)
typedef __attribute__((address_space(1))) v4i_vs GInt4;   // global int4
typedef __attribute__((address_space(3))) v4i_vs LInt4;   // LDS int4
#else
#define USE_ASYNC_LDS 0
#define WAIT_ASYNC(n) ((void)0)
#endif

__device__ __forceinline__ float fast_tanh(float v) {
#if __has_builtin(__builtin_amdgcn_tanhf)
    return __builtin_amdgcn_tanhf(v);           // v_tanh_f32 on gfx1250
#else
    return tanhf(v);
#endif
}

__device__ __forceinline__ v16h pack16(v8h lo, v8h hi) {
    v16h r;
#pragma unroll
    for (int j = 0; j < 8; ++j) { r[j] = lo[j]; r[j + 8] = hi[j]; }
    return r;
}

// Copy one thread's 64B slice of the B tile (contiguous in global AND LDS).
__device__ __forceinline__ void copy_b_tile(const _Float16* src, _Float16* dst) {
#if USE_ASYNC_LDS
    // INST_OFFSET is added to both the global and the LDS address (ISA §15.18),
    // and our 4 chunks are contiguous on both sides -> 4 async b128 loads.
    GInt4* g = (GInt4*)src;
    LInt4* l = (LInt4*)dst;
    __builtin_amdgcn_global_load_async_to_lds_b128(g, l, 0,  0);
    __builtin_amdgcn_global_load_async_to_lds_b128(g, l, 16, 0);
    __builtin_amdgcn_global_load_async_to_lds_b128(g, l, 32, 0);
    __builtin_amdgcn_global_load_async_to_lds_b128(g, l, 48, 0);
#else
    v8h b0 = *(const v8h*)(src);
    v8h b1 = *(const v8h*)(src + 8);
    v8h b2 = *(const v8h*)(src + 16);
    v8h b3 = *(const v8h*)(src + 24);
    *(v8h*)(dst)      = b0;
    *(v8h*)(dst + 8)  = b1;
    *(v8h*)(dst + 16) = b2;
    *(v8h*)(dst + 24) = b3;
#endif
}

// Gaussian basis producer: 4 x-values -> 32 f16 exp values into LDS.
__device__ __forceinline__ void produce_a(const float* xsrc, _Float16* adst) {
    const float invW  = 7.0f;                                // 1/width
    const float nhl2e = -0.5f * 1.44269504088896340736f;     // -0.5*log2(e)
    v4f xv = *(const v4f*)xsrc;
    __builtin_prefetch(xsrc + 32, 0, 3);                     // next cacheline of x
#pragma unroll
    for (int j = 0; j < 4; ++j) {
        const float t = fast_tanh(xv[j]);                    // zoom=1, pan=0
        v8h e;
#pragma unroll
        for (int g = 0; g < 8; ++g) {
            const float c = -1.0f + (2.0f / 7.0f) * (float)g;
            const float z = (t - c) * invW;
            e[g] = (_Float16)__builtin_amdgcn_exp2f(z * z * nhl2e);
        }
        *(v8h*)(adst + j * 8) = e;
    }
}

// ---------------------------------------------------------------------------
// Stage 1: W2[o, i*8+g] = (f16)(coeff[o,i,g] * weights[o,i])   (4 MB in d_ws)
// ---------------------------------------------------------------------------
__global__ __launch_bounds__(256) void w2_prep_kernel(
    const float* __restrict__ weights,    // [O, I]
    const float* __restrict__ coeff,      // [O, I, G]
    _Float16* __restrict__ w2)            // [O, K]
{
    const int idx = blockIdx.x * 256 + threadIdx.x;   // one (o,i) pair
    const float w = weights[idx];
    const float* c = coeff + (size_t)idx * GAUSS_G;
    v4f c0 = *(const v4f*)(c);
    v4f c1 = *(const v4f*)(c + 4);
    v8h r;
#pragma unroll
    for (int g = 0; g < 4; ++g) r[g]     = (_Float16)(c0[g] * w);
#pragma unroll
    for (int g = 0; g < 4; ++g) r[g + 4] = (_Float16)(c1[g] * w);
    *(v8h*)(w2 + (size_t)idx * GAUSS_G) = r;
}

// ---------------------------------------------------------------------------
// Stage 2: out = E @ W2^T, E computed on the fly, double-buffered LDS,
//          async B-tile prefetch (ASYNCcnt), f16 WMMA with f32 accum.
// grid = (BATCH/BM, OUT/BN), block = 256 (8 waves of 32)
// ---------------------------------------------------------------------------
__global__ __launch_bounds__(256) void gauss_gemm_kernel(
    const float* __restrict__ x,          // [BATCH, I]
    const _Float16* __restrict__ w2,      // [O, K]
    float* __restrict__ out)              // [BATCH, O]
{
    __shared__ _Float16 As[2][BM * LDS_STRIDE];   // 2 x 18 KB
    __shared__ _Float16 Bs[2][BN * LDS_STRIDE];   // 2 x 18 KB

    const int tid  = threadIdx.x;
    const int lane = tid & 31;
    const int wave = tid >> 5;
    const int wm   = wave & 3;            // 4 waves along M (32 rows each)
    const int wn   = wave >> 2;           // 2 waves along N (64 cols each)
    const int m0   = blockIdx.x * BM;
    const int n0   = blockIdx.y * BN;

    // producers: 2 threads per row, 64B (A: 4 x-values -> 32 f16) each
    const int prow  = tid >> 1;           // 0..127
    const int phalf = tid & 1;
    const float*    xrow  = x  + (size_t)(m0 + prow) * IN_SZ + phalf * 4;
    const _Float16* brow  = w2 + (size_t)(n0 + prow) * K_SZ  + phalf * 32;
    const int       poff  = prow * LDS_STRIDE + phalf * 32;

    v8f acc[2][4];
#pragma unroll
    for (int mi = 0; mi < 2; ++mi)
#pragma unroll
        for (int ni = 0; ni < 4; ++ni)
#pragma unroll
            for (int j = 0; j < 8; ++j) acc[mi][ni][j] = 0.0f;

    // ---- prologue: fill buffer 0 for kt = 0
    produce_a(xrow, &As[0][poff]);
    copy_b_tile(brow, &Bs[0][poff]);

    for (int kt = 0; kt < NT; ++kt) {
        const int cur = kt & 1;

        __syncthreads();                 // all reads of buf[cur^1] retired
        if (kt + 1 < NT) {
            const int nxt = cur ^ 1;
            copy_b_tile(brow + (kt + 1) * BK, &Bs[nxt][poff]);     // async
            produce_a(xrow + (kt + 1) * (BK / GAUSS_G), &As[nxt][poff]);
            WAIT_ASYNC(4);               // B(cur) done, B(kt+1) still in flight
        } else {
            WAIT_ASYNC(0);
        }
        __syncthreads();                 // publish buf[cur] to all waves

        const _Float16* Ab = As[cur];
        const _Float16* Bb = Bs[cur];

#pragma unroll
        for (int ks = 0; ks < 2; ++ks) {
            const int kk = ks * 32;

            // A frags: lane<16 -> K {0..7,16..23}; lane>=16 -> K {8..15,24..31}
            v16h a[2];
#pragma unroll
            for (int mi = 0; mi < 2; ++mi) {
                const _Float16* p = Ab + (wm * 32 + mi * 16 + (lane & 15)) * LDS_STRIDE
                                       + kk + ((lane >> 4) << 3);
                a[mi] = pack16(*(const v8h*)(p), *(const v8h*)(p + 16));
            }
            // B frags: lane<16 -> K {0..7,16..23} of col lane; lane>=16 -> K {8..15,24..31}
            v16h b[4];
#pragma unroll
            for (int ni = 0; ni < 4; ++ni) {
                const _Float16* p = Bb + (wn * 64 + ni * 16 + (lane & 15)) * LDS_STRIDE
                                       + kk + ((lane >> 4) << 3);
                b[ni] = pack16(*(const v8h*)(p), *(const v8h*)(p + 16));
            }
#pragma unroll
            for (int mi = 0; mi < 2; ++mi)
#pragma unroll
                for (int ni = 0; ni < 4; ++ni)
                    acc[mi][ni] = __builtin_amdgcn_wmma_f32_16x16x32_f16(
                        false, a[mi], false, b[ni],
                        (short)0, acc[mi][ni], false, false);
        }
    }

    // ---- write out: C VGPR j -> row j (lanes 0-15) / row j+8 (lanes 16-31)
#pragma unroll
    for (int mi = 0; mi < 2; ++mi) {
        const int rbase = m0 + wm * 32 + mi * 16 + ((lane >> 4) << 3);
#pragma unroll
        for (int ni = 0; ni < 4; ++ni) {
            const int col = n0 + wn * 64 + ni * 16 + (lane & 15);
#pragma unroll
            for (int j = 0; j < 8; ++j)
                out[(size_t)(rbase + j) * OUT_SZ + col] = acc[mi][ni][j];
        }
    }
}

// ---------------------------------------------------------------------------
extern "C" void kernel_launch(void* const* d_in, const int* in_sizes, int n_in,
                              void* d_out, int out_size, void* d_ws, size_t ws_size,
                              hipStream_t stream) {
    const float* x       = (const float*)d_in[0];   // [8192, 512]
    const float* weights = (const float*)d_in[1];   // [512, 512]
    const float* coeff   = (const float*)d_in[2];   // [512, 512, 8]
    float*       out     = (float*)d_out;           // [8192, 512]
    _Float16*    w2      = (_Float16*)d_ws;         // [512, 4096] f16, 4 MB

    w2_prep_kernel<<<(OUT_SZ * IN_SZ) / 256, 256, 0, stream>>>(weights, coeff, w2);

    dim3 grid(BATCH_SZ / BM, OUT_SZ / BN);
    gauss_gemm_kernel<<<grid, 256, 0, stream>>>(x, w2, out);
}
